// CTCPerSpeakerExtractorConcatNNG_63462436765739
// MI455X (gfx1250) — matrix operand order repaired and checked by reading, not compiled, not run on hardware
//
#include <hip/hip_runtime.h>
#include <hip/hip_bf16.h>
#include <math.h>

// ---------------- problem constants ----------------
#define BB       4
#define TT       1536
#define D_IN     512
#define DD       512
#define KSTREAMS 2
#define HH       8
#define DHD      64
#define BANDW    24
#define DFF      2048
#define MROWS    (BB * TT)          // 6144
#define TPAD     (TT + 64)          // padded time axis for vT (32 each side)
#define ALPHA_C  6.0f
#define BETA_C   0.5f
#define EPS_C    1e-5f

typedef unsigned short u16;
typedef __attribute__((ext_vector_type(16))) __bf16 v16bf;
typedef __attribute__((ext_vector_type(8)))  float  v8f;
typedef __attribute__((ext_vector_type(4)))  int    i32x4;

union Frag128 { uint4 q[2]; v16bf v; u16 s[16]; };

static __device__ __forceinline__ u16 f2bf(float f) {
  unsigned u = __float_as_uint(f);
  unsigned r = (u + 0x7FFFu + ((u >> 16) & 1u)) >> 16;
  return (u16)r;
}
static __device__ __forceinline__ float gelu_exact(float x) {
  return 0.5f * x * (1.0f + erff(x * 0.70710678118654752f));
}

// ---------- gfx1250 async global->LDS path (guarded; sync b128 fallback) ----------
#if defined(__gfx1250__) && __has_builtin(__builtin_amdgcn_global_load_async_to_lds_b128) && \
    __has_builtin(__builtin_amdgcn_s_wait_asynccnt)
#define USE_ASYNC_LDS 1
#else
#define USE_ASYNC_LDS 0
#endif

#if USE_ASYNC_LDS
typedef __attribute__((address_space(1))) i32x4 gi32x4;
typedef __attribute__((address_space(3))) i32x4 li32x4;
static __device__ __forceinline__ void stage_b128(const u16* g, u16* l) {
  __builtin_amdgcn_global_load_async_to_lds_b128((gi32x4*)g, (li32x4*)l, 0, 0);
}
static __device__ __forceinline__ void stage_wait() {          // all complete
  __builtin_amdgcn_s_wait_asynccnt(0);
}
static __device__ __forceinline__ void stage_wait_prev3() {    // allow 3 in flight
  __builtin_amdgcn_s_wait_asynccnt(3);
}
#else
static __device__ __forceinline__ void stage_b128(const u16* g, u16* l) {
  *(uint4*)l = *(const uint4*)g;
}
static __device__ __forceinline__ void stage_wait() {}
static __device__ __forceinline__ void stage_wait_prev3() {}
#endif

// ---------------- cast f32 -> bf16 (row-major) ----------------
__global__ void cast_bf16_kernel(const float* __restrict__ in, u16* __restrict__ out, int n) {
  int i = blockIdx.x * blockDim.x + threadIdx.x;
  int stride = gridDim.x * blockDim.x;
  for (; i < n; i += stride) out[i] = f2bf(in[i]);
}

// ---------------- cast f32 -> bf16 with transpose: in[K][N] -> out[N][K] ----------------
__global__ void cast_transpose_bf16_kernel(const float* __restrict__ in, u16* __restrict__ out,
                                           int K, int N) {
  int i = blockIdx.x * blockDim.x + threadIdx.x;
  int stride = gridDim.x * blockDim.x;
  int total = K * N;
  for (; i < total; i += stride) {
    int k = i / N, n = i % N;
    out[(size_t)n * K + k] = f2bf(in[i]);
  }
}

// ---------------- LayerNorm (+ optional sigmoid gate, + optional concat remap) ----------------
__global__ __launch_bounds__(256) void ln_kernel(
    const float* __restrict__ X, const float* __restrict__ Agate, int kk,
    const float* __restrict__ gam, const float* __restrict__ bet,
    u16*   __restrict__ out_bf, float* __restrict__ out_gated, float* __restrict__ outF)
{
  int row  = blockIdx.x * 8 + (threadIdx.x >> 5);
  int lane = threadIdx.x & 31;
  float w = 1.0f;
  if (Agate) {
    float a = Agate[(size_t)row * KSTREAMS + kk];
    w = 1.0f / (1.0f + expf(-ALPHA_C * (a - BETA_C)));
  }
  const float* src = X + (size_t)row * DD;
  float4 xv[4];
  float s = 0.f, s2 = 0.f;
#pragma unroll
  for (int i = 0; i < 4; ++i) {
    float4 v = *(const float4*)&src[(i * 32 + lane) * 4];
    v.x *= w; v.y *= w; v.z *= w; v.w *= w;
    xv[i] = v;
    s  += v.x + v.y + v.z + v.w;
    s2 += v.x * v.x + v.y * v.y + v.z * v.z + v.w * v.w;
  }
#pragma unroll
  for (int d = 1; d < 32; d <<= 1) { s += __shfl_xor(s, d); s2 += __shfl_xor(s2, d); }
  float mean = s * (1.0f / DD);
  float var  = s2 * (1.0f / DD) - mean * mean;
  float rinv = rsqrtf(var + EPS_C);
#pragma unroll
  for (int i = 0; i < 4; ++i) {
    int c0 = (i * 32 + lane) * 4;
    if (out_gated) *(float4*)&out_gated[(size_t)row * DD + c0] = xv[i];
    float4 g4 = *(const float4*)&gam[c0];
    float4 b4 = *(const float4*)&bet[c0];
    float4 y;
    y.x = (xv[i].x - mean) * rinv * g4.x + b4.x;
    y.y = (xv[i].y - mean) * rinv * g4.y + b4.y;
    y.z = (xv[i].z - mean) * rinv * g4.z + b4.z;
    y.w = (xv[i].w - mean) * rinv * g4.w + b4.w;
    if (out_bf) {
      ushort4 o; o.x = f2bf(y.x); o.y = f2bf(y.y); o.z = f2bf(y.z); o.w = f2bf(y.w);
      *(ushort4*)&out_bf[(size_t)row * DD + c0] = o;
    }
    if (outF) {
      int b = row / TT, t = row % TT;
      size_t orow = (size_t)b * (KSTREAMS * TT) + (size_t)kk * TT + t;
      *(float4*)&outF[orow * DD + c0] = y;
    }
  }
}

// ---------------- generic bf16 WMMA GEMM (weights pre-transposed) ----------------
// C[M,N] = act( A[M,K] @ W[K,N] + bias + bias2 ) (+ residual).  WT = W transposed [N][K].
// Block 256 thr = 8 waves; block tile 128(M) x 64(N); wave tile 32x32 (4 WMMAs).
// Double-buffered LDS; A & WT tiles stream via async global->LDS b128 overlapped with WMMA.
__global__ __launch_bounds__(256) void gemm_bf16_kernel(
    const u16* __restrict__ A, const u16* __restrict__ WT,
    const float* __restrict__ bias, const float* __restrict__ bias2,
    const float* __restrict__ resid,
    float* __restrict__ outF, u16* __restrict__ outBf, u16* __restrict__ outBfT,
    int Nn, int Kd, int actGelu)
{
  __shared__ __align__(16) u16 As[2][128][40];   // 2 x (128 m-rows x 32 k)
  __shared__ __align__(16) u16 Ws[2][64][40];    // 2 x (64 n-rows x 32 k)

  int tid  = threadIdx.x;
  int wave = tid >> 5, lane = tid & 31;
  int bm = blockIdx.y * 128, bn = blockIdx.x * 64;
  int wm = (wave & 3) * 32;
  int wn = (wave >> 2) * 32;

  int m  = lane & 15;
  int hf = lane >> 4;

  v8f acc00 = {}, acc01 = {}, acc10 = {}, acc11 = {};

  int arow = tid >> 1, acol = (tid & 1) * 16;   // A: 2 b128 per thread per tile
  int wrow = tid >> 2, wcol = (tid & 3) * 8;    // W: 1 b128 per thread per tile

  const u16* abase = A  + (size_t)(bm + arow) * Kd + acol;
  const u16* wbase = WT + (size_t)(bn + wrow) * Kd + wcol;

  auto stage_tile = [&](int k0, int buf) {
    stage_b128(abase + k0,     &As[buf][arow][acol]);
    stage_b128(abase + k0 + 8, &As[buf][arow][acol + 8]);
    stage_b128(wbase + k0,     &Ws[buf][wrow][wcol]);
  };

  int nk = Kd >> 5;                 // K-chunks of 32
  stage_tile(0, 0);
  for (int i = 0; i < nk; ++i) {
    int buf = i & 1;
    if (i + 1 < nk) {
      stage_tile((i + 1) << 5, buf ^ 1);       // overlap DMA of next tile with WMMA
      if (i + 2 < nk) {                        // prefetch tile i+2 into L2
        __builtin_prefetch(abase + ((i + 2) << 5), 0, 1);
        __builtin_prefetch(wbase + ((i + 2) << 5), 0, 1);
      }
      stage_wait_prev3();                      // tile i resident; next 3 still in flight
    } else {
      stage_wait();
    }
    __syncthreads();

    Frag128 a0, a1, b0, b1;
    a0.q[0] = *(const uint4*)&As[buf][wm + m][hf * 8];
    a0.q[1] = *(const uint4*)&As[buf][wm + m][hf * 8 + 16];
    a1.q[0] = *(const uint4*)&As[buf][wm + 16 + m][hf * 8];
    a1.q[1] = *(const uint4*)&As[buf][wm + 16 + m][hf * 8 + 16];
    b0.q[0] = *(const uint4*)&Ws[buf][wn + m][hf * 8];
    b0.q[1] = *(const uint4*)&Ws[buf][wn + m][hf * 8 + 16];
    b1.q[0] = *(const uint4*)&Ws[buf][wn + 16 + m][hf * 8];
    b1.q[1] = *(const uint4*)&Ws[buf][wn + 16 + m][hf * 8 + 16];
    acc00 = __builtin_amdgcn_wmma_f32_16x16x32_bf16(false, a0.v, false, b0.v,
                                                    (short)0, acc00, false, false);
    acc01 = __builtin_amdgcn_wmma_f32_16x16x32_bf16(false, a0.v, false, b1.v,
                                                    (short)0, acc01, false, false);
    acc10 = __builtin_amdgcn_wmma_f32_16x16x32_bf16(false, a1.v, false, b0.v,
                                                    (short)0, acc10, false, false);
    acc11 = __builtin_amdgcn_wmma_f32_16x16x32_bf16(false, a1.v, false, b1.v,
                                                    (short)0, acc11, false, false);
    __syncthreads();
  }

#pragma unroll
  for (int sm = 0; sm < 2; ++sm) {
#pragma unroll
    for (int sn = 0; sn < 2; ++sn) {
      const v8f& acc = sm ? (sn ? acc11 : acc10) : (sn ? acc01 : acc00);
      int gn = bn + wn + sn * 16 + m;
      float bs = bias ? bias[gn] : 0.0f;
      if (bias2) bs += bias2[gn];
#pragma unroll
      for (int vv = 0; vv < 8; ++vv) {
        int gm = bm + wm + sm * 16 + hf * 8 + vv;
        size_t idx = (size_t)gm * Nn + gn;
        float x = acc[vv] + bs;
        if (resid) x += resid[idx];
        if (actGelu) x = gelu_exact(x);
        if (outF)  outF[idx]  = x;
        if (outBf) outBf[idx] = f2bf(x);
        if (outBfT) {                          // transposed, time-padded store (for V)
          int bcur = gm / TT, t = gm % TT;
          outBfT[((size_t)bcur * DD + gn) * TPAD + 32 + t] = f2bf(x);
        }
      }
    }
  }
}

// ---------------- banded attention (BAND=24 -> 64-wide window per 16-row tile) ----------------
// q,k row-major bf16 [B*T,512]; vT transposed+padded bf16 [B*512][TPAD] (offset 32).
// Out-of-band / out-of-range columns are masked in softmax, so clamped/padded reads are safe.
__global__ __launch_bounds__(32) void band_attn_kernel(
    const u16* __restrict__ q, const u16* __restrict__ k,
    const u16* __restrict__ vT, u16* __restrict__ y)
{
  __shared__ __align__(16) u16 qs[16][64];
  __shared__ __align__(16) u16 ks[64][72];   // [key][dh]
  __shared__ __align__(16) u16 vs[64][72];   // [dh][key] straight from vT
  __shared__ __align__(16) u16 ps[16][64];

  int tile = blockIdx.x, h = blockIdx.y, b = blockIdx.z;
  int lane = threadIdx.x;

  int r0 = tile * 16;
  int c0 = r0 - BANDW;                        // 64-wide window = 16 + 2*24

  {   // q tile: 2 lanes/row, 4 x b128 each (async)
    int row = lane >> 1, cb = (lane & 1) * 32;
    const u16* src = q + ((size_t)(b * TT + r0 + row)) * DD + h * DHD + cb;
#pragma unroll
    for (int c4 = 0; c4 < 4; ++c4)
      stage_b128(src + c4 * 8, &qs[row][cb + c4 * 8]);
  }
#pragma unroll
  for (int rr = 0; rr < 2; ++rr) {            // k window rows (clamped; OOB masked later)
    int row = lane * 2 + rr;
    int t = c0 + row;
    int tc = t < 0 ? 0 : (t >= TT ? TT - 1 : t);
    const u16* sk = k + ((size_t)(b * TT + tc)) * DD + h * DHD;
#pragma unroll
    for (int c8 = 0; c8 < 8; ++c8)
      stage_b128(sk + c8 * 8, &ks[row][c8 * 8]);
  }
#pragma unroll
  for (int rr = 0; rr < 2; ++rr) {            // v window rows from vT: contiguous (padded)
    int dh = lane * 2 + rr;
    const u16* sv = vT + ((size_t)(b * DD + h * DHD + dh)) * TPAD + 32 + c0;
#pragma unroll
    for (int c8 = 0; c8 < 8; ++c8)
      stage_b128(sv + c8 * 8, &vs[dh][c8 * 8]);
  }
  stage_wait();
  __syncthreads();

  int m  = lane & 15;
  int hf = lane >> 4;

  // scores = Q(16x64) @ K^T -> 4 n-tiles, DH split in 2 K-chunks of 32
  float sc[4][8];
#pragma unroll
  for (int nt = 0; nt < 4; ++nt) {
    v8f acc = {};
#pragma unroll
    for (int kc = 0; kc < 2; ++kc) {
      Frag128 af, bf;
      af.q[0] = *(const uint4*)&qs[m][kc * 32 + hf * 8];
      af.q[1] = *(const uint4*)&qs[m][kc * 32 + hf * 8 + 16];
      bf.q[0] = *(const uint4*)&ks[nt * 16 + m][kc * 32 + hf * 8];
      bf.q[1] = *(const uint4*)&ks[nt * 16 + m][kc * 32 + hf * 8 + 16];
      acc = __builtin_amdgcn_wmma_f32_16x16x32_bf16(false, af.v, false, bf.v,
                                                    (short)0, acc, false, false);
    }
#pragma unroll
    for (int vv = 0; vv < 8; ++vv) sc[nt][vv] = acc[vv];
  }

  // masked softmax (C layout: col = lane&15, rows = hf*8+vv; 16 lanes share a row)
#pragma unroll
  for (int vv = 0; vv < 8; ++vv) {
    int mrow = hf * 8 + vv;
    float val[4];
    float mx = -3.0e38f;
#pragma unroll
    for (int nt = 0; nt < 4; ++nt) {
      int jj = nt * 16 + m;
      int gj = c0 + jj;
      bool ok = (jj >= mrow) && (jj <= mrow + 2 * BANDW) && (gj >= 0) && (gj < TT);
      val[nt] = ok ? sc[nt][vv] * 0.125f : -3.0e38f;   // 1/sqrt(64)
      mx = fmaxf(mx, val[nt]);
    }
#pragma unroll
    for (int d = 1; d < 16; d <<= 1) mx = fmaxf(mx, __shfl_xor(mx, d));
    float s = 0.f;
#pragma unroll
    for (int nt = 0; nt < 4; ++nt) { val[nt] = expf(val[nt] - mx); s += val[nt]; }
#pragma unroll
    for (int d = 1; d < 16; d <<= 1) s += __shfl_xor(s, d);
    float inv = 1.0f / s;
#pragma unroll
    for (int nt = 0; nt < 4; ++nt) ps[mrow][nt * 16 + m] = f2bf(val[nt] * inv);
  }
  __syncthreads();

  // y = P(16x64) @ V(64x64): B-fragments contiguous from vs[dh][key]
#pragma unroll
  for (int ntd = 0; ntd < 4; ++ntd) {
    v8f acc = {};
#pragma unroll
    for (int kc = 0; kc < 2; ++kc) {
      Frag128 af, bf;
      af.q[0] = *(const uint4*)&ps[m][kc * 32 + hf * 8];
      af.q[1] = *(const uint4*)&ps[m][kc * 32 + hf * 8 + 16];
      bf.q[0] = *(const uint4*)&vs[ntd * 16 + m][kc * 32 + hf * 8];
      bf.q[1] = *(const uint4*)&vs[ntd * 16 + m][kc * 32 + hf * 8 + 16];
      acc = __builtin_amdgcn_wmma_f32_16x16x32_bf16(false, af.v, false, bf.v,
                                                    (short)0, acc, false, false);
    }
#pragma unroll
    for (int vv = 0; vv < 8; ++vv) {
      int gm = r0 + hf * 8 + vv;
      y[((size_t)(b * TT + gm)) * DD + h * DHD + ntd * 16 + m] = f2bf(acc[vv]);
    }
  }
}

// ---------------- host-side orchestration ----------------
extern "C" void kernel_launch(void* const* d_in, const int* in_sizes, int n_in,
                              void* d_out, int out_size, void* d_ws, size_t ws_size,
                              hipStream_t stream) {
  const float* x_m    = (const float*)d_in[0];
  const float* A      = (const float*)d_in[1];
  const float* W_in   = (const float*)d_in[2];
  const float* b_in   = (const float*)d_in[3];
  const float* ln_q_g = (const float*)d_in[4];
  const float* ln_q_b = (const float*)d_in[5];
  const float* ln_kv_g= (const float*)d_in[6];
  const float* ln_kv_b= (const float*)d_in[7];
  const float* Wq     = (const float*)d_in[8];
  const float* bq     = (const float*)d_in[9];
  const float* Wk     = (const float*)d_in[10];
  const float* bk     = (const float*)d_in[11];
  const float* Wv     = (const float*)d_in[12];
  const float* bv     = (const float*)d_in[13];
  const float* Wo     = (const float*)d_in[14];
  const float* bo     = (const float*)d_in[15];
  const float* ln_f_g = (const float*)d_in[16];
  const float* ln_f_b = (const float*)d_in[17];
  const float* W1     = (const float*)d_in[18];
  const float* b1     = (const float*)d_in[19];
  const float* W2     = (const float*)d_in[20];
  const float* b2     = (const float*)d_in[21];
  const float* ln_s_g = (const float*)d_in[22];
  const float* ln_s_b = (const float*)d_in[23];
  const float* spk    = (const float*)d_in[24];
  float* out = (float*)d_out;

  char* base = (char*)d_ws;
  size_t off = 0;
  auto take = [&](size_t bytes) { char* p = base + off; off += (bytes + 255) & ~(size_t)255; return p; };
  u16*   xm_bf   = (u16*)  take((size_t)MROWS * D_IN * 2);
  u16*   WinT_bf = (u16*)  take((size_t)D_IN * DD * 2);      // [DD][D_IN]
  u16*   WqT_bf  = (u16*)  take((size_t)DD * DD * 2);
  u16*   WkT_bf  = (u16*)  take((size_t)DD * DD * 2);
  u16*   WvT_bf  = (u16*)  take((size_t)DD * DD * 2);
  u16*   WoT_bf  = (u16*)  take((size_t)DD * DD * 2);
  u16*   W1T_bf  = (u16*)  take((size_t)DD * DFF * 2);       // [DFF][DD]
  u16*   W2T_bf  = (u16*)  take((size_t)DFF * DD * 2);       // [DD][DFF]
  float* X_f32   = (float*)take((size_t)MROWS * DD * 4);
  u16*   KVln_bf = (u16*)  take((size_t)MROWS * DD * 2);
  u16*   k_bf    = (u16*)  take((size_t)MROWS * DD * 2);
  u16*   vT_bf   = (u16*)  take((size_t)BB * DD * TPAD * 2); // transposed, padded V
  float* Xk_f32  = (float*)take((size_t)MROWS * DD * 4);
  u16*   Qln_bf  = (u16*)  take((size_t)MROWS * DD * 2);
  u16*   q_bf    = (u16*)  take((size_t)MROWS * DD * 2);
  u16*   attn_bf = (u16*)  take((size_t)MROWS * DD * 2);
  float* y_f32   = (float*)take((size_t)MROWS * DD * 4);
  u16*   hln_bf  = (u16*)  take((size_t)MROWS * DD * 2);
  u16*   ffn1_bf = (u16*)  take((size_t)MROWS * DFF * 2);
  float* pre_f32 = (float*)take((size_t)MROWS * DD * 4);
  (void)ws_size; (void)in_sizes; (void)n_in; (void)out_size;

  cast_bf16_kernel<<<(MROWS * D_IN + 1023) / 1024, 256, 0, stream>>>(x_m, xm_bf, MROWS * D_IN);
  auto castT = [&](const float* s, u16* d, int K, int N) {
    cast_transpose_bf16_kernel<<<(K * N + 1023) / 1024, 256, 0, stream>>>(s, d, K, N);
  };
  castT(W_in, WinT_bf, D_IN, DD);
  castT(Wq, WqT_bf, DD, DD);  castT(Wk, WkT_bf, DD, DD);
  castT(Wv, WvT_bf, DD, DD);  castT(Wo, WoT_bf, DD, DD);
  castT(W1, W1T_bf, DD, DFF); castT(W2, W2T_bf, DFF, DD);

  dim3 gB(256);
  dim3 gridD(DD / 64, MROWS / 128);    // N=512
  dim3 gridF(DFF / 64, MROWS / 128);   // N=2048
  dim3 lnGrid(MROWS / 8);

  gemm_bf16_kernel<<<gridD, gB, 0, stream>>>(xm_bf, WinT_bf, b_in, nullptr, nullptr,
                                             X_f32, nullptr, nullptr, DD, D_IN, 0);
  ln_kernel<<<lnGrid, gB, 0, stream>>>(X_f32, nullptr, 0, ln_kv_g, ln_kv_b,
                                       KVln_bf, nullptr, nullptr);
  gemm_bf16_kernel<<<gridD, gB, 0, stream>>>(KVln_bf, WkT_bf, bk, nullptr, nullptr,
                                             nullptr, k_bf, nullptr, DD, DD, 0);
  gemm_bf16_kernel<<<gridD, gB, 0, stream>>>(KVln_bf, WvT_bf, bv, nullptr, nullptr,
                                             nullptr, nullptr, vT_bf, DD, DD, 0);

  for (int kk = 0; kk < KSTREAMS; ++kk) {
    ln_kernel<<<lnGrid, gB, 0, stream>>>(X_f32, A, kk, ln_q_g, ln_q_b,
                                         Qln_bf, Xk_f32, nullptr);
    gemm_bf16_kernel<<<gridD, gB, 0, stream>>>(Qln_bf, WqT_bf, bq, nullptr, nullptr,
                                               nullptr, q_bf, nullptr, DD, DD, 0);
    band_attn_kernel<<<dim3(TT / 16, HH, BB), 32, 0, stream>>>(q_bf, k_bf, vT_bf, attn_bf);
    gemm_bf16_kernel<<<gridD, gB, 0, stream>>>(attn_bf, WoT_bf, bo, nullptr, Xk_f32,
                                               y_f32, nullptr, nullptr, DD, DD, 0);
    ln_kernel<<<lnGrid, gB, 0, stream>>>(y_f32, nullptr, 0, ln_f_g, ln_f_b,
                                         hln_bf, nullptr, nullptr);
    gemm_bf16_kernel<<<gridF, gB, 0, stream>>>(hln_bf, W1T_bf, b1, nullptr, nullptr,
                                               nullptr, ffn1_bf, nullptr, DFF, DD, 1);
    gemm_bf16_kernel<<<gridD, gB, 0, stream>>>(ffn1_bf, W2T_bf, b2, spk + (size_t)kk * DD,
                                               y_f32, pre_f32, nullptr, nullptr, DD, DFF, 0);
    ln_kernel<<<lnGrid, gB, 0, stream>>>(pre_f32, nullptr, kk, ln_s_g, ln_s_b,
                                         nullptr, nullptr, out);
  }
}
